// DualGraphGNN_79534204388003
// MI455X (gfx1250) — compile-verified, matrix operand order checked by reading, not run on hardware
//
#include <hip/hip_runtime.h>
#include <hip/hip_bf16.h>
#include <math.h>

// ---------------------------------------------------------------------------
// DualGraphGNN on MI455X (gfx1250, wave32).
// GEMMs use V_WMMA_F32_16X16X4_F32 (full fp32 precision, matrix pipe) when the
// builtin exists; otherwise fall back to the codegen-confirmed
// V_WMMA_F32_16X16X32_F16 with bounds-checked f32->f16 fragment fills.
// All other stages (edge softmax, segment ops, layernorm) are scalar kernels.
// ---------------------------------------------------------------------------

#define HID    128
#define EMB    64
#define HEADS  4
#define NLAYERS 3
#define EDGEDIM 4
#define A_IN   92
#define P_IN   32
#define N_A    16384
#define E_A    196608
#define N_P    4096
#define E_P    32768
#define NBATCH 64
#define EA_TOT (E_A + N_A)   /* self-loops appended */

typedef float    v2f  __attribute__((ext_vector_type(2)));
typedef float    v8f  __attribute__((ext_vector_type(8)));
typedef _Float16 v16h __attribute__((ext_vector_type(16)));

#if __has_builtin(__builtin_amdgcn_wmma_f32_16x16x4_f32)
#define USE_WMMA_F32X4 1
#else
#define USE_WMMA_F32X4 0
#endif

// ---------------------------------------------------------------------------
// Generic GEMM: C[M,N] = act(A[M,K] @ B[K,N] + bias), row-major, lda=K, ldb=N.
// One wave32 computes one 16x16 tile. M,N must be multiples of 16; K mult. of 4.
// Early return is wave-uniform so EXEC is all-ones at every WMMA.
// ---------------------------------------------------------------------------
__global__ void gemm_wmma(const float* __restrict__ A, const float* __restrict__ B,
                          const float* __restrict__ bias, float* __restrict__ C,
                          int M, int N, int K, int ldc, int relu) {
  const int wave  = (blockIdx.x * blockDim.x + threadIdx.x) >> 5;
  const int lane  = threadIdx.x & 31;
  const int tilesN = N >> 4;
  const int total  = (M >> 4) * tilesN;
  if (wave >= total) return;                     // wave-uniform
  const int tm = wave / tilesN;
  const int tn = wave - tm * tilesN;
  const int half = lane >> 4;                    // 0: lanes 0-15, 1: lanes 16-31
  const int lid  = lane & 15;
  const int arow = tm * 16 + lid;                // A row for this lane's A-frag
  const int bcol = tn * 16 + lid;                // global output column
  v8f acc = {0.f, 0.f, 0.f, 0.f, 0.f, 0.f, 0.f, 0.f};

#if USE_WMMA_F32X4
  // f32 A 16x4 frag: lanes0-15 hold K={k,k+1}, lanes16-31 hold K={k+2,k+3}.
  // f32 B 4x16 frag: same K split, N striped across lanes.
  for (int k = 0; k < K; k += 4) {
    __builtin_prefetch(&A[(size_t)arow * K + k + 16], 0, 0);
    v2f a = *(const v2f*)(A + (size_t)arow * K + k + 2 * half);  // contiguous pair
    v2f b;
    b.x = B[(size_t)(k + 2 * half)     * N + bcol];
    b.y = B[(size_t)(k + 2 * half + 1) * N + bcol];
    acc = __builtin_amdgcn_wmma_f32_16x16x4_f32(
        false, a, false, b, (short)0, acc, false, false);
  }
#else
  // Fallback: f16 inputs, f32 accumulate, K stepped by 32 with zero-fill tail.
  for (int k0 = 0; k0 < K; k0 += 32) {
    v16h af, bf;
#pragma unroll
    for (int v = 0; v < 8; ++v) {
      const int kbase = (v < 4) ? (2 * v) : (16 + 2 * (v - 4));
#pragma unroll
      for (int j = 0; j < 2; ++j) {
        const int kk = k0 + kbase + 8 * half + j;
        const float av = (kk < K) ? A[(size_t)arow * K + kk] : 0.f;
        const float bv = (kk < K) ? B[(size_t)kk * N + bcol] : 0.f;
        af[2 * v + j] = (_Float16)av;
        bf[2 * v + j] = (_Float16)bv;
      }
    }
    acc = __builtin_amdgcn_wmma_f32_16x16x32_f16(
        false, af, false, bf, (short)0, acc, false, false);
  }
#endif

  const float bv = bias ? bias[bcol] : 0.f;
#pragma unroll
  for (int r = 0; r < 8; ++r) {
    const int row = tm * 16 + r + 8 * half;      // C/D layout: VGPR r -> M=r(+8)
    float v = acc[r] + bv;
    if (relu) v = fmaxf(v, 0.f);
    C[(size_t)row * ldc + bcol] = v;
  }
}

// ---------------------------------------------------------------------------
__global__ void fill_f32(float* __restrict__ p, float v, int n) {
  int t = blockIdx.x * blockDim.x + threadIdx.x;
  if (t < n) p[t] = v;
}

__device__ __forceinline__ void atomicMaxF(float* addr, float val) {
  int* ia = (int*)addr;
  int old = __float_as_int(*addr);
  while (__int_as_float(old) < val) {
    int assumed = old;
    old = atomicCAS(ia, assumed, __float_as_int(val));
    if (old == assumed) break;
  }
}

__device__ __forceinline__ float lrelu02(float x) { return x > 0.f ? x : 0.2f * x; }

// one wave per (node, head): a_s/a_d = sum_c x[n,h,c]*att_{src,dst}[h,c]
__global__ void att_scores(const float* __restrict__ x,
                           const float* __restrict__ att_src,
                           const float* __restrict__ att_dst,
                           float* __restrict__ a_s, float* __restrict__ a_d, int N) {
  const int gw = (blockIdx.x * blockDim.x + threadIdx.x) >> 5;
  const int lane = threadIdx.x & 31;
  if (gw >= N * HEADS) return;
  const int n = gw >> 2, h = gw & 3;
  const float* xr = x + (size_t)n * (HEADS * HID) + h * HID;
  const float* ws = att_src + h * HID;
  const float* wd = att_dst + h * HID;
  float ss = 0.f, sd = 0.f;
#pragma unroll
  for (int j = 0; j < 4; ++j) {
    const float xv = xr[lane + 32 * j];
    ss += xv * ws[lane + 32 * j];
    sd += xv * wd[lane + 32 * j];
  }
  for (int off = 16; off > 0; off >>= 1) {
    ss += __shfl_down(ss, off, 32);
    sd += __shfl_down(sd, off, 32);
  }
  if (lane == 0) { a_s[n * HEADS + h] = ss; a_d[n * HEADS + h] = sd; }
}

__device__ __forceinline__ void edge_sd(const int* __restrict__ src,
                                        const int* __restrict__ dst,
                                        int e, int& s, int& d) {
  if (e < E_A) { s = src[e]; d = dst[e]; }
  else         { s = d = e - E_A; }              // appended self-loop
}

__global__ void gat_edge_max(const int* __restrict__ src, const int* __restrict__ dst,
                             const float* __restrict__ a_s, const float* __restrict__ a_d,
                             float* __restrict__ amax) {
  const int e = blockIdx.x * blockDim.x + threadIdx.x;
  if (e >= EA_TOT) return;
  int s, d; edge_sd(src, dst, e, s, d);
#pragma unroll
  for (int h = 0; h < HEADS; ++h) {
    const float al = lrelu02(a_s[s * HEADS + h] + a_d[d * HEADS + h]);
    atomicMaxF(&amax[d * HEADS + h], al);
  }
}

__global__ void gat_edge_expsum(const int* __restrict__ src, const int* __restrict__ dst,
                                const float* __restrict__ a_s, const float* __restrict__ a_d,
                                const float* __restrict__ amax, float* __restrict__ denom) {
  const int e = blockIdx.x * blockDim.x + threadIdx.x;
  if (e >= EA_TOT) return;
  int s, d; edge_sd(src, dst, e, s, d);
#pragma unroll
  for (int h = 0; h < HEADS; ++h) {
    const float al = lrelu02(a_s[s * HEADS + h] + a_d[d * HEADS + h]);
    atomicAdd(&denom[d * HEADS + h], __expf(al - amax[d * HEADS + h]));
  }
}

// one wave per edge; lane covers 4 channels; head-mean folded in (0.25 factor)
__global__ void gat_edge_accum(const int* __restrict__ src, const int* __restrict__ dst,
                               const float* __restrict__ x,
                               const float* __restrict__ a_s, const float* __restrict__ a_d,
                               const float* __restrict__ amax, const float* __restrict__ denom,
                               float* __restrict__ acc) {
  const int gw = (blockIdx.x * blockDim.x + threadIdx.x) >> 5;
  const int lane = threadIdx.x & 31;
  if (gw >= EA_TOT) return;
  int s, d; edge_sd(src, dst, gw, s, d);
  float w[HEADS];
#pragma unroll
  for (int h = 0; h < HEADS; ++h) {
    const float al = lrelu02(a_s[s * HEADS + h] + a_d[d * HEADS + h]);
    const float ev = __expf(al - amax[d * HEADS + h]);
    w[h] = ev / (denom[d * HEADS + h] + 1e-16f) * 0.25f;
  }
#pragma unroll
  for (int j = 0; j < 4; ++j) {
    const int c = lane + 32 * j;
    float v = 0.f;
#pragma unroll
    for (int h = 0; h < HEADS; ++h)
      v += x[(size_t)s * (HEADS * HID) + h * HID + c] * w[h];
    atomicAdd(&acc[(size_t)d * HID + c], v);
  }
}

// block of 128 per node: h = relu(LN(acc + gbias)) (+ old h if residual)
__global__ void node_finalize(const float* __restrict__ acc, const float* __restrict__ gbias,
                              const float* __restrict__ lng, const float* __restrict__ lnb,
                              float* __restrict__ h, int residual) {
  const int n = blockIdx.x, c = threadIdx.x;
  __shared__ float red[HID];
  float v = acc[(size_t)n * HID + c] + (gbias ? gbias[c] : 0.f);
  red[c] = v; __syncthreads();
  for (int s = 64; s > 0; s >>= 1) { if (c < s) red[c] += red[c + s]; __syncthreads(); }
  const float mean = red[0] * (1.f / HID); __syncthreads();
  const float dv = v - mean;
  red[c] = dv * dv; __syncthreads();
  for (int s = 64; s > 0; s >>= 1) { if (c < s) red[c] += red[c + s]; __syncthreads(); }
  const float var = red[0] * (1.f / HID);
  float y = dv * rsqrtf(var + 1e-5f) * lng[c] + lnb[c];
  y = fmaxf(y, 0.f);
  h[(size_t)n * HID + c] = residual ? (y + h[(size_t)n * HID + c]) : y;
}

__global__ void pool_accum(const float* __restrict__ h, const int* __restrict__ batch,
                           float* __restrict__ pool, float* __restrict__ cnt, int N) {
  const int t = blockIdx.x * blockDim.x + threadIdx.x;
  if (t >= N * HID) return;
  const int n = t >> 7, c = t & 127;
  const int b = batch[n];
  atomicAdd(&pool[(size_t)b * HID + c], h[t]);
  if (c == 0) atomicAdd(&cnt[b], 1.f);
}

__global__ void pool_div(float* __restrict__ pool, const float* __restrict__ cnt) {
  const int t = blockIdx.x * blockDim.x + threadIdx.x;
  if (t >= NBATCH * HID) return;
  pool[t] /= fmaxf(cnt[t >> 7], 1.f);
}

// mi[e] = concat(h[row], h[col], edge_attr[e])  -> [E_P, 260]
__global__ void build_mi(const float* __restrict__ h, const int* __restrict__ row,
                         const int* __restrict__ col, const float* __restrict__ ea,
                         float* __restrict__ mi) {
  const int K = 2 * HID + EDGEDIM;
  const int t = blockIdx.x * blockDim.x + threadIdx.x;
  if (t >= E_P * K) return;
  const int e = t / K, c = t - e * K;
  float v;
  if (c < HID)            v = h[(size_t)row[e] * HID + c];
  else if (c < 2 * HID)   v = h[(size_t)col[e] * HID + (c - HID)];
  else                    v = ea[(size_t)e * EDGEDIM + (c - 2 * HID)];
  mi[t] = v;
}

__global__ void aggr_sum(const float* __restrict__ m, const int* __restrict__ row,
                         float* __restrict__ aggr) {
  const int t = blockIdx.x * blockDim.x + threadIdx.x;
  if (t >= E_P * HID) return;
  const int e = t >> 7, c = t & 127;
  atomicAdd(&aggr[(size_t)row[e] * HID + c], m[t]);
}

__global__ void build_ui(const float* __restrict__ h, const float* __restrict__ aggr,
                         float* __restrict__ ui) {
  const int t = blockIdx.x * blockDim.x + threadIdx.x;
  if (t >= N_P * 2 * HID) return;
  const int n = t >> 8, c = t & 255;
  ui[t] = (c < HID) ? h[(size_t)n * HID + c] : aggr[(size_t)n * HID + (c - HID)];
}

// final head l3: out[b] = dot(z2[b,:64], w) + bias
__global__ void head_l3(const float* __restrict__ z2, const float* __restrict__ w,
                        const float* __restrict__ b, float* __restrict__ out) {
  const int t = threadIdx.x;
  if (t >= NBATCH) return;
  float s = 0.f;
#pragma unroll
  for (int k = 0; k < EMB; ++k) s += z2[(size_t)t * EMB + k] * w[k];
  out[t] = s + b[0];
}

// ---------------------------------------------------------------------------
static inline int cdiv(long a, long b) { return (int)((a + b - 1) / b); }

static void launch_gemm(const float* A, const float* B, const float* bias, float* C,
                        int M, int N, int K, int ldc, int relu, hipStream_t s) {
  const long threads = (long)(M / 16) * (N / 16) * 32;
  gemm_wmma<<<cdiv(threads, 256), 256, 0, s>>>(A, B, bias, C, M, N, K, ldc, relu);
}
static void fill(float* p, float v, long n, hipStream_t s) {
  fill_f32<<<cdiv(n, 256), 256, 0, s>>>(p, v, (int)n);
}

extern "C" void kernel_launch(void* const* d_in, const int* in_sizes, int n_in,
                              void* d_out, int out_size, void* d_ws, size_t ws_size,
                              hipStream_t stream) {
  // --- input map (depth-first insertion-order flatten of setup_inputs()) ---
  const float* ax      = (const float*)d_in[0];   // [N_A, 92]
  const float* px      = (const float*)d_in[1];   // [N_P, 32]
  const float* p_eattr = (const float*)d_in[2];   // [E_P, 4]
  const float* a_inp_w = (const float*)d_in[3];   // [92,128]
  const float* a_inp_b = (const float*)d_in[4];
  // a gat layers l: 5+4l..8+4l = lin_w[128,512], att_src[4,128], att_dst, bias[128]
  // a ln layers  l: 17+2l, 18+2l
  const float* a_out_w = (const float*)d_in[23];  // [128,64]
  const float* a_out_b = (const float*)d_in[24];
  const float* p_inp_w = (const float*)d_in[25];  // [32,128]
  const float* p_inp_b = (const float*)d_in[26];
  // p ecc layers l: 27+8l..34+8l = m1_w[260,128],m1_b,m2_w[128,128],m2_b,
  //                                u1_w[256,128],u1_b,u2_w[128,128],u2_b
  // p ln layers  l: 51+2l, 52+2l
  const float* p_out_w = (const float*)d_in[57];  // [128,64]
  const float* p_out_b = (const float*)d_in[58];
  const float* h_l1_w  = (const float*)d_in[59];  // [128,128]
  const float* h_l1_b  = (const float*)d_in[60];
  const float* h_l2_w  = (const float*)d_in[61];  // [128,64]
  const float* h_l2_b  = (const float*)d_in[62];
  const float* h_l3_w  = (const float*)d_in[63];  // [64,1]
  const float* h_l3_b  = (const float*)d_in[64];
  const int* a_ei      = (const int*)d_in[65];    // [2, E_A]
  const int* a_batch   = (const int*)d_in[66];
  const int* p_ei      = (const int*)d_in[67];    // [2, E_P]
  const int* p_batch   = (const int*)d_in[68];
  const int* a_src = a_ei;         const int* a_dst = a_ei + E_A;
  const int* p_row = p_ei;         const int* p_col = p_ei + E_P;

  float* ws = (float*)d_ws;
  // --- persistent small region ---
  float* z    = ws + 0;                 // [64,128]  concat(ae,pe)
  float* z1   = ws + 64 * 128;          // [64,128]
  float* z2   = ws + 2 * 64 * 128;      // [64,64]
  float* pool = ws + 2 * 64 * 128 + 64 * 64;  // [64,128]
  float* cnt  = pool + 64 * 128;              // [64]
  const size_t ARENA = 32768;
  float* arena = ws + ARENA;

  // ======================= ATOM GNN (GAT x3) =======================
  float* a_h    = arena;                              // [N_A,128]
  float* a_x    = a_h + (size_t)N_A * HID;            // [N_A,512]
  float* a_as   = a_x + (size_t)N_A * HEADS * HID;    // [N_A,4]
  float* a_ad   = a_as + (size_t)N_A * HEADS;
  float* a_amax = a_ad + (size_t)N_A * HEADS;
  float* a_den  = a_amax + (size_t)N_A * HEADS;
  float* a_acc  = a_den + (size_t)N_A * HEADS;        // [N_A,128]

  launch_gemm(ax, a_inp_w, a_inp_b, a_h, N_A, HID, A_IN, HID, 1, stream);
  for (int l = 0; l < NLAYERS; ++l) {
    const float* lin_w   = (const float*)d_in[5 + 4 * l];
    const float* att_src = (const float*)d_in[6 + 4 * l];
    const float* att_dst = (const float*)d_in[7 + 4 * l];
    const float* gbias   = (const float*)d_in[8 + 4 * l];
    const float* lng     = (const float*)d_in[17 + 2 * l];
    const float* lnb     = (const float*)d_in[18 + 2 * l];
    launch_gemm(a_h, lin_w, nullptr, a_x, N_A, HEADS * HID, HID, HEADS * HID, 0, stream);
    att_scores<<<cdiv((long)N_A * HEADS * 32, 256), 256, 0, stream>>>(
        a_x, att_src, att_dst, a_as, a_ad, N_A);
    fill(a_amax, -1e30f, (long)N_A * HEADS, stream);
    fill(a_den, 0.f, (long)N_A * HEADS, stream);
    fill(a_acc, 0.f, (long)N_A * HID, stream);
    gat_edge_max<<<cdiv(EA_TOT, 256), 256, 0, stream>>>(a_src, a_dst, a_as, a_ad, a_amax);
    gat_edge_expsum<<<cdiv(EA_TOT, 256), 256, 0, stream>>>(a_src, a_dst, a_as, a_ad, a_amax, a_den);
    gat_edge_accum<<<cdiv((long)EA_TOT * 32, 256), 256, 0, stream>>>(
        a_src, a_dst, a_x, a_as, a_ad, a_amax, a_den, a_acc);
    node_finalize<<<N_A, HID, 0, stream>>>(a_acc, gbias, lng, lnb, a_h, l > 0);
  }
  fill(pool, 0.f, NBATCH * HID, stream);
  fill(cnt, 0.f, NBATCH, stream);
  pool_accum<<<cdiv((long)N_A * HID, 256), 256, 0, stream>>>(a_h, a_batch, pool, cnt, N_A);
  pool_div<<<cdiv(NBATCH * HID, 256), 256, 0, stream>>>(pool, cnt);
  launch_gemm(pool, a_out_w, a_out_b, z /*cols 0..63*/, NBATCH, EMB, HID, 2 * EMB, 0, stream);

  // ======================= POLY GNN (ECC x3) =======================
  // atom arena fully drained into z; reuse arena
  float* p_h  = arena;                                // [N_P,128]
  float* mi   = p_h + (size_t)N_P * HID;              // [E_P,260]
  float* m_t  = mi + (size_t)E_P * (2 * HID + EDGEDIM); // [E_P,128]
  float* m    = mi;                                   // reuse mi region for m
  float* aggr = m_t + (size_t)E_P * HID;              // [N_P,128]
  float* ui   = aggr + (size_t)N_P * HID;             // [N_P,256]
  float* tbuf = ui + (size_t)N_P * 2 * HID;           // [N_P,128]
  float* hn   = tbuf + (size_t)N_P * HID;             // [N_P,128]

  launch_gemm(px, p_inp_w, p_inp_b, p_h, N_P, HID, P_IN, HID, 1, stream);
  for (int l = 0; l < NLAYERS; ++l) {
    const int base = 27 + 8 * l;
    const float* m1_w = (const float*)d_in[base + 0];
    const float* m1_b = (const float*)d_in[base + 1];
    const float* m2_w = (const float*)d_in[base + 2];
    const float* m2_b = (const float*)d_in[base + 3];
    const float* u1_w = (const float*)d_in[base + 4];
    const float* u1_b = (const float*)d_in[base + 5];
    const float* u2_w = (const float*)d_in[base + 6];
    const float* u2_b = (const float*)d_in[base + 7];
    const float* lng  = (const float*)d_in[51 + 2 * l];
    const float* lnb  = (const float*)d_in[52 + 2 * l];
    build_mi<<<cdiv((long)E_P * (2 * HID + EDGEDIM), 256), 256, 0, stream>>>(
        p_h, p_row, p_col, p_eattr, mi);
    launch_gemm(mi, m1_w, m1_b, m_t, E_P, HID, 2 * HID + EDGEDIM, HID, 1, stream);
    launch_gemm(m_t, m2_w, m2_b, m, E_P, HID, HID, HID, 0, stream);   // m overwrites mi
    fill(aggr, 0.f, (long)N_P * HID, stream);
    aggr_sum<<<cdiv((long)E_P * HID, 256), 256, 0, stream>>>(m, p_row, aggr);
    build_ui<<<cdiv((long)N_P * 2 * HID, 256), 256, 0, stream>>>(p_h, aggr, ui);
    launch_gemm(ui, u1_w, u1_b, tbuf, N_P, HID, 2 * HID, HID, 1, stream);
    launch_gemm(tbuf, u2_w, u2_b, hn, N_P, HID, HID, HID, 0, stream);
    node_finalize<<<N_P, HID, 0, stream>>>(hn, nullptr, lng, lnb, p_h, l > 0);
  }
  fill(pool, 0.f, NBATCH * HID, stream);
  fill(cnt, 0.f, NBATCH, stream);
  pool_accum<<<cdiv((long)N_P * HID, 256), 256, 0, stream>>>(p_h, p_batch, pool, cnt, N_P);
  pool_div<<<cdiv(NBATCH * HID, 256), 256, 0, stream>>>(pool, cnt);
  launch_gemm(pool, p_out_w, p_out_b, z + EMB /*cols 64..127*/, NBATCH, EMB, HID, 2 * EMB, 0, stream);

  // ======================= HEAD MLP =======================
  launch_gemm(z, h_l1_w, h_l1_b, z1, NBATCH, HID, 2 * EMB, HID, 1, stream);
  launch_gemm(z1, h_l2_w, h_l2_b, z2, NBATCH, EMB, HID, EMB, 1, stream);
  head_l3<<<1, 64, 0, stream>>>(z2, h_l3_w, h_l3_b, (float*)d_out);
}
